// LinearFusionAttention_17145509446111
// MI455X (gfx1250) — matrix operand order repaired and checked by reading, not compiled, not run on hardware
//
#include <hip/hip_runtime.h>
#include <hip/hip_bf16.h>
#include <math.h>

// ---------------------------------------------------------------------------
// LinearFusionAttention for MI455X (gfx1250, wave32, WMMA)
// B=8 H=64 W=64 C=384 NH=12 HD=32 N=4096
// ---------------------------------------------------------------------------

#define Bc   8
#define Hc   64
#define Wc   64
#define Cc   384
#define NHc  12
#define HDc  32
#define Nc   4096
#define BN   (Bc * Nc)          // 32768 rows
#define C2   (2 * Cc)           // 768 projection cols

typedef __attribute__((ext_vector_type(16))) __bf16 v16bf;
typedef __attribute__((ext_vector_type(8)))  __bf16 v8bf;
typedef __attribute__((ext_vector_type(8)))  float  v8f;

// log(10000)/96
#define ROPE_LOGK 0.09594104551f

// ===========================================================================
// Kernel 0: elementwise f32 -> bf16 (8 elems/thread, vectorized)
// ===========================================================================
__global__ __launch_bounds__(256)
void cvt_bf16_kernel(const float* __restrict__ in, __bf16* __restrict__ outp,
                     int n) {
    const int i = (blockIdx.x * blockDim.x + threadIdx.x) * 8;
    if (i >= n) return;
    const float4 a = ((const float4*)(in + i))[0];
    const float4 b = ((const float4*)(in + i))[1];
    v8bf o;
    o[0] = (__bf16)a.x; o[1] = (__bf16)a.y;
    o[2] = (__bf16)a.z; o[3] = (__bf16)a.w;
    o[4] = (__bf16)b.x; o[5] = (__bf16)b.y;
    o[6] = (__bf16)b.z; o[7] = (__bf16)b.w;
    *(v8bf*)(outp + i) = o;
}

// ===========================================================================
// Kernel 1: qk = elu1(X @ W^T + b); emit q_rope/k_rope (bf16, RoPE applied),
//           q_plain (f32), and atomic partial sums of k (for mean).
// Wave tile: 32 (M=tokens) x 64 (N=cols), K=384 in steps of 32.
// A/B operands pre-converted to bf16 -> pure b128 fragment loads, no converts
// in the hot loop.
// ===========================================================================
__global__ __launch_bounds__(256)
void proj_rope_kernel(const __bf16* __restrict__ Xb,   // (BN, C) bf16
                      const __bf16* __restrict__ Wb,   // (2C, C) bf16
                      const float*  __restrict__ bias, // (2C)
                      __bf16* __restrict__ q_rope,     // (BN, C) bf16
                      __bf16* __restrict__ k_rope,     // (BN, C) bf16
                      float*  __restrict__ q_plain,    // (BN, C) f32
                      float*  __restrict__ kmean)      // (B, C) f32 sums
{
    const int lane  = threadIdx.x & 31;
    const int wave  = threadIdx.x >> 5;
    const int wtile = blockIdx.x * 8 + wave;      // 12288 wave tiles
    const int mtile = wtile / (C2 / 64);          // 0..1023
    const int ntile = wtile % (C2 / 64);          // 0..11
    const int m0 = mtile * 32;
    const int n0 = ntile * 64;
    const bool is_k = (n0 >= Cc);                 // ntiles 6..11 are the K half

    const int lm = lane & 15;
    const int lh = lane >> 4;

    v8f acc[2][4];
#pragma unroll
    for (int mt = 0; mt < 2; ++mt)
#pragma unroll
        for (int nt = 0; nt < 4; ++nt)
#pragma unroll
            for (int r = 0; r < 8; ++r) acc[mt][nt][r] = 0.0f;

    for (int k0 = 0; k0 < Cc; k0 += 32) {
        // ---- A fragments: 16x32 bf16, M = token row, K = input channel ----
        v16bf afrag[2];
#pragma unroll
        for (int mt = 0; mt < 2; ++mt) {
            const __bf16* row = Xb + (size_t)(m0 + mt * 16 + lm) * Cc + k0;
            const v8bf lo = *(const v8bf*)(row + lh * 16);
            const v8bf hi = *(const v8bf*)(row + lh * 16 + 8);
            afrag[mt] = __builtin_shufflevector(lo, hi,
                0, 1, 2, 3, 4, 5, 6, 7, 8, 9, 10, 11, 12, 13, 14, 15);
        }
        // ---- B fragments: 32x16 bf16, B[k][n] = Wb[n][k] -------------------
        v16bf bfrag[4];
#pragma unroll
        for (int nt = 0; nt < 4; ++nt)
            bfrag[nt] = *(const v16bf*)(
                Wb + (size_t)(n0 + nt * 16 + lm) * Cc + k0 + lh * 16);
#pragma unroll
        for (int mt = 0; mt < 2; ++mt)
#pragma unroll
            for (int nt = 0; nt < 4; ++nt)
                acc[mt][nt] = __builtin_amdgcn_wmma_f32_16x16x32_bf16(
                    false, afrag[mt], false, bfrag[nt],
                    (short)0, acc[mt][nt], false, false);
    }

    // ---- Epilogue: bias + elu1 + RoPE + stores ----------------------------
#pragma unroll
    for (int nt = 0; nt < 4; ++nt) {
        const int ng = n0 + nt * 16 + lm;            // global projection col
        const int c  = is_k ? (ng - Cc) : ng;        // channel within q or k
        const float bsv = bias[ng];
        const int p  = c >> 1;                       // rope pair index
        const int pm = (p < 96) ? p : (p - 96);
        const float theta = __expf(-(float)pm * ROPE_LOGK);
        float ksum = 0.0f;
#pragma unroll
        for (int mt = 0; mt < 2; ++mt) {
#pragma unroll
            for (int r = 0; r < 8; ++r) {
                float v = acc[mt][nt][r] + bsv;
                v = (v > 0.0f) ? (v + 1.0f) : __expf(v);   // elu(x)+1
                const int token = m0 + mt * 16 + r + (lh ? 8 : 0);
                const int nrow  = token & (Nc - 1);
                const int hi = nrow >> 6, wi = nrow & 63;
                const float pos = (p < 96) ? (float)hi : (float)wi;
                float sa, ca;
                __sincosf(pos * theta, &sa, &ca);
                const float pv = __shfl_xor(v, 1);          // rope pair partner
                const float rv = (c & 1) ? (sa * pv + ca * v)
                                         : (ca * v - sa * pv);
                const int addr = token * Cc + c;
                if (is_k) {
                    k_rope[addr] = (__bf16)rv;
                    ksum += v;
                } else {
                    q_rope[addr]  = (__bf16)rv;
                    q_plain[addr] = v;
                }
            }
        }
        if (is_k) {
            const int b = m0 >> 12;                  // 4096 tokens per batch
            atomicAdd(&kmean[b * Cc + c], ksum);
        }
    }
}

// ===========================================================================
// Kernel 2: kv[b,h] = (1/N) * k_rope[b,h]^T @ v[b,h]   (32x32 per head)
// 8 waves per (b,h), each reduces 512 tokens with 2x2 WMMA tiles; f32 atomics.
// ===========================================================================
__global__ __launch_bounds__(128)
void kv_kernel(const __bf16* __restrict__ k_rope,   // (BN, C) bf16
               const __bf16* __restrict__ Vb,       // (BN, C) bf16 (raw input)
               float* __restrict__ kv)              // (B*NH, 32, 32)
{
    const int lane = threadIdx.x & 31;
    const int wave = threadIdx.x >> 5;
    const int w    = blockIdx.x * 4 + wave;         // 0..767
    const int head = w >> 3;                        // 0..95
    const int seg  = w & 7;
    const int b = head / NHc, h = head % NHc;
    const int tokbase = b * Nc + seg * 512;
    const int c0 = h * HDc;
    const int lm = lane & 15, lh = lane >> 4;

    v8f acc[2][2];
#pragma unroll
    for (int dt = 0; dt < 2; ++dt)
#pragma unroll
        for (int et = 0; et < 2; ++et)
#pragma unroll
            for (int r = 0; r < 8; ++r) acc[dt][et][r] = 0.0f;

    for (int t0 = 0; t0 < 512; t0 += 32) {
        // A: 16x32, M = d (head dim), K = token -> strided gather of bf16
        v16bf afrag[2];
#pragma unroll
        for (int dt = 0; dt < 2; ++dt) {
            const __bf16* col =
                k_rope + (size_t)(tokbase + t0) * Cc + c0 + dt * 16 + lm;
#pragma unroll
            for (int j = 0; j < 8; ++j)
                afrag[dt][j]     = col[(size_t)(lh * 16 + j) * Cc];
#pragma unroll
            for (int j = 0; j < 8; ++j)
                afrag[dt][8 + j] = col[(size_t)(lh * 16 + 8 + j) * Cc];
        }
        // B: 32x16, K = token, N = e (head dim) -> strided gather of bf16
        v16bf bfrag[2];
#pragma unroll
        for (int et = 0; et < 2; ++et) {
            const __bf16* col =
                Vb + (size_t)(tokbase + t0) * Cc + c0 + et * 16 + lm;
#pragma unroll
            for (int j = 0; j < 16; ++j)
                bfrag[et][j] = col[(size_t)(lh * 16 + j) * Cc];
        }
#pragma unroll
        for (int dt = 0; dt < 2; ++dt)
#pragma unroll
            for (int et = 0; et < 2; ++et)
                acc[dt][et] = __builtin_amdgcn_wmma_f32_16x16x32_bf16(
                    false, afrag[dt], false, bfrag[et],
                    (short)0, acc[dt][et], false, false);
    }

    const float scale = 1.0f / (float)Nc;
    float* kvh = kv + head * (HDc * HDc);
#pragma unroll
    for (int dt = 0; dt < 2; ++dt)
#pragma unroll
        for (int et = 0; et < 2; ++et)
#pragma unroll
            for (int r = 0; r < 8; ++r) {
                const int d = dt * 16 + r + (lh ? 8 : 0);
                const int e = et * 16 + lm;
                atomicAdd(&kvh[d * HDc + e], acc[dt][et][r] * scale);
            }
}

// ===========================================================================
// Kernel 3: z[b,h,n] = 1 / (dot(q_plain[b,n,h,:], kmean[b,h,:]) / N + 1e-6)
// ===========================================================================
__global__ __launch_bounds__(256)
void z_kernel(const float* __restrict__ q_plain,    // other stream's q
              const float* __restrict__ kmean,      // (B, C) sums of k
              float* __restrict__ z)                // (B*NH, N)
{
    const int gid = blockIdx.x * blockDim.x + threadIdx.x;
    if (gid >= Bc * NHc * Nc) return;
    const int n  = gid & (Nc - 1);
    const int bh = gid >> 12;
    const int b = bh / NHc, h = bh % NHc;
    const float* q  = q_plain + (size_t)(b * Nc + n) * Cc + h * HDc;
    const float* km = kmean + b * Cc + h * HDc;
    float dot = 0.0f;
#pragma unroll
    for (int d = 0; d < HDc; ++d) dot += q[d] * km[d];
    dot *= (1.0f / (float)Nc);
    z[gid] = 1.0f / (dot + 1e-6f);
}

// ===========================================================================
// Kernel 4: out[b,n,c] = z[b,h,n] * sum_d q_rope[b,n,h,d]*kv[b,h,d,e]
//                        + dwconv3x3(Xin)[b,n,c]
// ===========================================================================
__global__ __launch_bounds__(256)
void out_kernel(const __bf16* __restrict__ q_rope,  // other stream's q_rope
                const float*  __restrict__ kv,      // (B*NH, 32, 32)
                const float*  __restrict__ z,       // (B*NH, N)
                const float*  __restrict__ Xin,     // same-stream input (f32)
                const float*  __restrict__ wl,      // (C, 9) dw weights
                const float*  __restrict__ bl,      // (C) dw bias
                float* __restrict__ out)            // (BN, C)
{
    const int gid = blockIdx.x * blockDim.x + threadIdx.x;
    if (gid >= BN * Cc) return;
    const int c     = gid % Cc;
    const int token = gid / Cc;
    const int b    = token >> 12;
    const int nrow = token & (Nc - 1);
    const int h = c >> 5, e = c & 31;

    // ---- linear attention readout -----------------------------------------
    const __bf16* q  = q_rope + (size_t)token * Cc + h * HDc;
    const float* kvc = kv + (b * NHc + h) * (HDc * HDc) + e;
    float attn = 0.0f;
#pragma unroll
    for (int d = 0; d < HDc; ++d) attn += (float)q[d] * kvc[d * HDc];
    attn *= z[(b * NHc + h) * Nc + nrow];

    // ---- depthwise 3x3 conv (SAME, zero pad) ------------------------------
    const int hi = nrow >> 6, wi = nrow & 63;
    float dw = bl[c];
    const float* wc = wl + c * 9;
#pragma unroll
    for (int dy = 0; dy < 3; ++dy) {
#pragma unroll
        for (int dx = 0; dx < 3; ++dx) {
            const int hh = hi + dy - 1;
            const int ww = wi + dx - 1;
            if (hh >= 0 && hh < Hc && ww >= 0 && ww < Wc)
                dw += Xin[((size_t)(b << 12) + hh * Wc + ww) * Cc + c] *
                      wc[dy * 3 + dx];
        }
    }
    out[gid] = attn + dw;
}

// ===========================================================================
// Host launcher
// ===========================================================================
extern "C" void kernel_launch(void* const* d_in, const int* in_sizes, int n_in,
                              void* d_out, int out_size, void* d_ws, size_t ws_size,
                              hipStream_t stream) {
    const float* x       = (const float*)d_in[0];
    const float* y       = (const float*)d_in[1];
    const float* w_qk_l  = (const float*)d_in[2];
    const float* b_qk_l  = (const float*)d_in[3];
    const float* w_qk_i  = (const float*)d_in[4];
    const float* b_qk_i  = (const float*)d_in[5];
    const float* w_lepe_l = (const float*)d_in[6];
    const float* b_lepe_l = (const float*)d_in[7];
    const float* w_lepe_i = (const float*)d_in[8];
    const float* b_lepe_i = (const float*)d_in[9];
    float* out = (float*)d_out;

    // ---- workspace carve-up ------------------------------------------------
    char* ws = (char*)d_ws;
    const size_t SZ_KMEAN = (size_t)Bc * Cc * sizeof(float);        //  12 KB
    const size_t SZ_KV    = (size_t)Bc * NHc * HDc * HDc * sizeof(float); // 384 KB
    const size_t SZ_BF16  = (size_t)BN * Cc * sizeof(__bf16);       //  24 MB
    const size_t SZ_F32   = (size_t)BN * Cc * sizeof(float);        //  48 MB
    const size_t SZ_Z     = (size_t)Bc * NHc * Nc * sizeof(float);  // 1.5 MB
    const size_t SZ_WBF   = (size_t)C2 * Cc * sizeof(__bf16);       // 576 KB

    size_t o = 0;
    float*  kmean_l  = (float*)(ws + o);  o += SZ_KMEAN;
    float*  kmean_i  = (float*)(ws + o);  o += SZ_KMEAN;
    float*  kv_l     = (float*)(ws + o);  o += SZ_KV;
    float*  kv_i     = (float*)(ws + o);  o += SZ_KV;
    const size_t zero_bytes = o;          // contiguous atomic targets
    __bf16* q_rope_l = (__bf16*)(ws + o); o += SZ_BF16;
    __bf16* k_rope_l = (__bf16*)(ws + o); o += SZ_BF16;
    __bf16* q_rope_i = (__bf16*)(ws + o); o += SZ_BF16;
    __bf16* k_rope_i = (__bf16*)(ws + o); o += SZ_BF16;
    float*  q_plain_l = (float*)(ws + o); o += SZ_F32;
    float*  q_plain_i = (float*)(ws + o); o += SZ_F32;
    float*  z_l      = (float*)(ws + o);  o += SZ_Z;
    float*  z_i      = (float*)(ws + o);  o += SZ_Z;
    __bf16* x_bf     = (__bf16*)(ws + o); o += SZ_BF16;
    __bf16* y_bf     = (__bf16*)(ws + o); o += SZ_BF16;
    __bf16* wl_bf    = (__bf16*)(ws + o); o += SZ_WBF;
    __bf16* wi_bf    = (__bf16*)(ws + o); o += SZ_WBF;

    hipMemsetAsync(d_ws, 0, zero_bytes, stream);

    // ---- one-shot bf16 conversion of GEMM operands ------------------------
    const int nX = BN * Cc;        // 12,582,912 (divisible by 2048)
    const int nW = C2 * Cc;        // 294,912
    cvt_bf16_kernel<<<nX / (256 * 8), 256, 0, stream>>>(x, x_bf, nX);
    cvt_bf16_kernel<<<nX / (256 * 8), 256, 0, stream>>>(y, y_bf, nX);
    cvt_bf16_kernel<<<nW / (256 * 8), 256, 0, stream>>>(w_qk_l, wl_bf, nW);
    cvt_bf16_kernel<<<nW / (256 * 8), 256, 0, stream>>>(w_qk_i, wi_bf, nW);

    // ---- projections (WMMA GEMM + elu1 + RoPE) ----------------------------
    const int proj_blocks = (BN / 32) * (C2 / 64) / 8;   // 1536
    proj_rope_kernel<<<proj_blocks, 256, 0, stream>>>(
        x_bf, wl_bf, b_qk_l, q_rope_l, k_rope_l, q_plain_l, kmean_l);
    proj_rope_kernel<<<proj_blocks, 256, 0, stream>>>(
        y_bf, wi_bf, b_qk_i, q_rope_i, k_rope_i, q_plain_i, kmean_i);

    // ---- kv reductions (WMMA) ---------------------------------------------
    kv_kernel<<<192, 128, 0, stream>>>(k_rope_l, x_bf, kv_l);
    kv_kernel<<<192, 128, 0, stream>>>(k_rope_i, y_bf, kv_i);

    // ---- z normalizers ----------------------------------------------------
    const int z_blocks = (Bc * NHc * Nc + 255) / 256;    // 1536
    z_kernel<<<z_blocks, 256, 0, stream>>>(q_plain_i, kmean_l, z_l);
    z_kernel<<<z_blocks, 256, 0, stream>>>(q_plain_l, kmean_i, z_i);

    // ---- outputs: attention readout + depthwise conv ----------------------
    const int out_blocks = (BN * Cc + 255) / 256;        // 49152
    out_kernel<<<out_blocks, 256, 0, stream>>>(
        q_rope_i, kv_l, z_l, x, w_lepe_l, b_lepe_l, out);
    out_kernel<<<out_blocks, 256, 0, stream>>>(
        q_rope_l, kv_i, z_i, y, w_lepe_i, b_lepe_i, out + (size_t)BN * Cc);
}
